// Model_90426241450476
// MI455X (gfx1250) — compile-verified, matrix operand order checked by reading, not run on hardware
//
#include <hip/hip_runtime.h>
#include <hip/hip_bf16.h>
#include <math.h>

// ---------------------------------------------------------------------------
// CDNA5 (gfx1250) implementation.
//   N = 65536, D = 2, C = 1, PDE = 1, DE = 5, E = 11, K = 16, LAT = 64
// Heavy GEMMs (dist-transform 64x64 / 64x704, head 128x64 / 64x32) use
// v_wmma_f32_16x16x32_f16 (wave32, 16x16 f32 accum in 8 VGPRs).
// Activations use a branch-free fast tanh (v_exp_f32 + v_rcp_f32).
// ---------------------------------------------------------------------------

typedef __attribute__((ext_vector_type(16))) _Float16 v16h;
typedef __attribute__((ext_vector_type(8)))  float    v8f;

#define NKNN 16
#define LAT 64
#define EDIM 11

static __device__ __forceinline__ v8f wmma_f16(v16h a, v16h b, v8f c) {
    return __builtin_amdgcn_wmma_f32_16x16x32_f16(
        /*neg_a=*/false, a, /*neg_b=*/false, b,
        /*c_mod=*/(short)0, c, /*reuse_a=*/false, /*reuse_b=*/false);
}

// Branch-free tanh: tanh(x) = sign(x) * (1 - t) / (1 + t), t = exp(-2|x|).
// Lowers to v_exp_f32 + v_rcp_f32 + a few FMAs; no EXEC divergence.
static __device__ __forceinline__ float fast_tanh(float x) {
    float ax = fabsf(x);
    float t  = __expf(-2.0f * ax);
    float r  = __builtin_amdgcn_rcpf(1.0f + t);
    float y  = (1.0f - t) * r;
    return copysignf(y, x);
}

// A fragment (16x32, f16) from f32 row-major source.
// ISA 7.12.2: lanes 0-15 hold M=lane with K = kb+{0..7,16..23};
//             lanes 16-31 hold M=lane-16 with K = kb+{8..15,24..31}.
static __device__ __forceinline__ v16h fragA_f32(const float* __restrict__ base,
                                                 int ld, int kb, int lane) {
    int m   = lane & 15;
    int klo = (lane & 16) ? 8 : 0;
    const float* p = base + (size_t)m * ld + kb + klo;
    v16h a;
#pragma unroll
    for (int h = 0; h < 8; ++h) {
        a[h]     = (_Float16)p[h];
        a[h + 8] = (_Float16)p[h + 16];
    }
    return a;
}

// A fragment from f16 row-major LDS tile.
static __device__ __forceinline__ v16h fragA_f16(const _Float16* base,
                                                 int ld, int kb, int lane) {
    int m   = lane & 15;
    int klo = (lane & 16) ? 8 : 0;
    const _Float16* p = base + (size_t)m * ld + kb + klo;
    v16h a;
#pragma unroll
    for (int h = 0; h < 8; ++h) {
        a[h]     = p[h];
        a[h + 8] = p[h + 16];
    }
    return a;
}

// B fragment (32x16, f16) from K-contiguous transposed weights Bt[n][k].
static __device__ __forceinline__ v16h fragB(const _Float16* __restrict__ bt,
                                             int ldk, int n0, int kb, int lane) {
    int n   = n0 + (lane & 15);
    int klo = (lane & 16) ? 8 : 0;
    const _Float16* p = bt + (size_t)n * ldk + kb + klo;
    v16h b;
#pragma unroll
    for (int h = 0; h < 8; ++h) {
        b[h]     = p[h];
        b[h + 8] = p[h + 16];
    }
    return b;
}

// ---------------------------------------------------------------------------
// Kernel 0: zero the latent accumulator.
// ---------------------------------------------------------------------------
__global__ void init_kernel(float* latent_sum) {
    latent_sum[threadIdx.x] = 0.0f;
}

// ---------------------------------------------------------------------------
// Kernel 1: per-node lt MLP (12->16->32->64, tanh everywhere), sum into
// latent_sum (mean done later).
// ---------------------------------------------------------------------------
__global__ __launch_bounds__(256) void latent_kernel(
    const float* __restrict__ means, const float* __restrict__ cov,
    const float* __restrict__ u, const float* __restrict__ su,
    const float* __restrict__ sux, const float* __restrict__ suxx,
    const float* __restrict__ W1, const float* __restrict__ b1,
    const float* __restrict__ W2, const float* __restrict__ b2,
    const float* __restrict__ W3, const float* __restrict__ b3,
    float* __restrict__ latent_sum, int N)
{
    __shared__ float s[64];
    if (threadIdx.x < 64) s[threadIdx.x] = 0.0f;
    __syncthreads();

    float acc[64];
#pragma unroll
    for (int j = 0; j < 64; ++j) acc[j] = 0.0f;

    for (int n = blockIdx.x * blockDim.x + threadIdx.x; n < N;
         n += gridDim.x * blockDim.x) {
        float x[12];
        x[0] = means[n * 2 + 0]; x[1] = means[n * 2 + 1];
        x[2] = cov[n * 4 + 0]; x[3] = cov[n * 4 + 1];
        x[4] = cov[n * 4 + 2]; x[5] = cov[n * 4 + 3];
        x[6] = u[n]; x[7] = su[n];
        x[8] = sux[n * 2 + 0]; x[9] = sux[n * 2 + 1];
        x[10] = suxx[n * 2 + 0]; x[11] = suxx[n * 2 + 1];

        float h1[16];
#pragma unroll
        for (int j = 0; j < 16; ++j) {
            float a = b1[j];
            for (int i = 0; i < 12; ++i) a += x[i] * W1[i * 16 + j];
            h1[j] = fast_tanh(a);
        }
        float h2[32];
#pragma unroll
        for (int j = 0; j < 32; ++j) {
            float a = b2[j];
            for (int i = 0; i < 16; ++i) a += h1[i] * W2[i * 32 + j];
            h2[j] = fast_tanh(a);
        }
#pragma unroll
        for (int j = 0; j < 64; ++j) {
            float a = b3[j];
            for (int i = 0; i < 32; ++i) a += h2[i] * W3[i * 64 + j];
            acc[j] += fast_tanh(a);
        }
    }
#pragma unroll
    for (int j = 0; j < 64; ++j) atomicAdd(&s[j], acc[j]);
    __syncthreads();
    if (threadIdx.x < 64) atomicAdd(&latent_sum[threadIdx.x], s[threadIdx.x]);
}

// ---------------------------------------------------------------------------
// Kernel 2: the five tiny T-nets (64->32->32->dd^2), T = I + out.
// Tm layout: T[4] @0, Tu[1] @4, Tux[4] @5, Tuxx[4] @9, Tpde[1] @13.
// ---------------------------------------------------------------------------
struct TnetArgs {
    const float* W1[5]; const float* b1[5];
    const float* W2[5]; const float* b2[5];
    const float* W3[5]; const float* b3[5];
    int dd[5]; int ooff[5];
};

__global__ void tnet_kernel(const float* __restrict__ latent_sum, float invN,
                            TnetArgs ta, float* __restrict__ Tm)
{
    __shared__ float lat[64], h1[32], h2[32];
    int t = threadIdx.x;
    if (t < 64) lat[t] = latent_sum[t] * invN;
    __syncthreads();
    for (int s = 0; s < 5; ++s) {
        if (t < 32) {
            float a = ta.b1[s][t];
            for (int i = 0; i < 64; ++i) a += lat[i] * ta.W1[s][i * 32 + t];
            h1[t] = fast_tanh(a);
        }
        __syncthreads();
        if (t < 32) {
            float a = ta.b2[s][t];
            for (int i = 0; i < 32; ++i) a += h1[i] * ta.W2[s][i * 32 + t];
            h2[t] = fast_tanh(a);
        }
        __syncthreads();
        int dd = ta.dd[s], d2 = dd * dd;
        if (t < d2) {
            float a = ta.b3[s][t];
            for (int i = 0; i < 32; ++i) a += h2[i] * ta.W3[s][i * d2 + t];
            Tm[ta.ooff[s] + t] = a + (((t / dd) == (t % dd)) ? 1.0f : 0.0f);
        }
        __syncthreads();
    }
}

// ---------------------------------------------------------------------------
// Kernel 3: t_params (11) + ip MLP (11->16->32->64) -> feats.
// ---------------------------------------------------------------------------
__global__ __launch_bounds__(256) void feats_kernel(
    const float* __restrict__ cov, const float* __restrict__ u,
    const float* __restrict__ su, const float* __restrict__ sux,
    const float* __restrict__ suxx, const float* __restrict__ pde,
    const float* __restrict__ Tm,
    const float* __restrict__ W1, const float* __restrict__ b1,
    const float* __restrict__ W2, const float* __restrict__ b2,
    const float* __restrict__ W3, const float* __restrict__ b3,
    float* __restrict__ feats, int N)
{
    int n = blockIdx.x * blockDim.x + threadIdx.x;
    if (n >= N) return;

    float T00 = Tm[0], T01 = Tm[1], T10 = Tm[2], T11 = Tm[3];
    float Tu = Tm[4];
    const float* cv = cov + (size_t)n * 4;

    float x[11];
    x[0] = T00 * cv[0] + T01 * cv[2];
    x[1] = T00 * cv[1] + T01 * cv[3];
    x[2] = T10 * cv[0] + T11 * cv[2];
    x[3] = T10 * cv[1] + T11 * cv[3];
    x[4] = u[n] * Tu;
    x[5] = su[n] * Tu;
    float ux0 = sux[n * 2 + 0], ux1 = sux[n * 2 + 1];
    x[6] = Tm[5] * ux0 + Tm[6] * ux1;
    x[7] = Tm[7] * ux0 + Tm[8] * ux1;
    float xx0 = suxx[n * 2 + 0], xx1 = suxx[n * 2 + 1];
    x[8] = Tm[9]  * xx0 + Tm[10] * xx1;
    x[9] = Tm[11] * xx0 + Tm[12] * xx1;
    x[10] = pde[n] * Tm[13];

    float h1[16];
#pragma unroll
    for (int j = 0; j < 16; ++j) {
        float a = b1[j];
        for (int i = 0; i < 11; ++i) a += x[i] * W1[i * 16 + j];
        h1[j] = fast_tanh(a);
    }
    float h2[32];
#pragma unroll
    for (int j = 0; j < 32; ++j) {
        float a = b2[j];
        for (int i = 0; i < 16; ++i) a += h1[i] * W2[i * 32 + j];
        h2[j] = fast_tanh(a);
    }
#pragma unroll
    for (int j = 0; j < 64; ++j) {
        float a = b3[j];
        for (int i = 0; i < 32; ++i) a += h2[i] * W3[i * 64 + j];
        feats[(size_t)n * 64 + j] = a;
    }
}

// ---------------------------------------------------------------------------
// Kernel 4: transpose+convert WMMA weights to f16 (K-contiguous Bt[n][k]).
// ---------------------------------------------------------------------------
__global__ void convert_weights_kernel(
    const float* __restrict__ dtfW1, const float* __restrict__ dtfW2,
    const float* __restrict__ dnW1,  const float* __restrict__ dnW2,
    _Float16* __restrict__ Bt_dtf1, _Float16* __restrict__ Bt_dtf2,
    _Float16* __restrict__ Bt_dn1,  _Float16* __restrict__ Bt_dn2)
{
    int stride = gridDim.x * blockDim.x;
    int t = blockIdx.x * blockDim.x + threadIdx.x;
    for (int i = t; i < 64 * 64; i += stride) {           // dtf.W1 (64,64)
        int n = i / 64, k = i % 64;
        Bt_dtf1[i] = (_Float16)dtfW1[k * 64 + n];
    }
    for (int i = t; i < 704 * 64; i += stride) {          // dtf.W2 (64,704)
        int n = i / 64, k = i % 64;
        Bt_dtf2[i] = (_Float16)dtfW2[k * 704 + n];
    }
    for (int i = t; i < 64 * 128; i += stride) {          // dn.W1 (128,64)
        int n = i / 128, k = i % 128;
        Bt_dn1[i] = (_Float16)dnW1[k * 64 + n];
    }
    for (int i = t; i < 32 * 64; i += stride) {           // dn.W2 (64,32)
        int n = i / 64, k = i % 64;
        Bt_dn2[i] = (_Float16)dnW2[k * 32 + n];
    }
}

// ---------------------------------------------------------------------------
// Kernel 5 (fused, WMMA): per 16-node tile:
//   H  = tanh(X @ dtfW1 + b1)            16x64   (waves 0..3, WMMA)
//   Tr = H @ dtfW2 + b2                  16x704  (8 waves, WMMA) -> LDS
//   nf[n,l] = mean_k ( (Tr[n,l,:] . emb[n,k,:]) * X[nbr[n,k], l] )
// ---------------------------------------------------------------------------
__global__ __launch_bounds__(256) void aggregate_kernel(
    const float* __restrict__ Xin, const float* __restrict__ means,
    const int* __restrict__ nbr,
    const _Float16* __restrict__ Bt1, const float* __restrict__ b1,
    const _Float16* __restrict__ Bt2, const float* __restrict__ b2,
    float* __restrict__ nfOut)
{
    __shared__ float    Tr[16 * 704];          // 45056 B
    __shared__ _Float16 Hh[16 * 64];           //  2048 B
    __shared__ float    Emb[16 * NKNN * EDIM]; // 11264 B
    __shared__ int      Nb[16 * NKNN];         //  1024 B

    int row0 = blockIdx.x * 16;
    int t = threadIdx.x;
    int lane = t & 31, wave = t >> 5;

    // stage 0: neighbor indices + positional embeddings (all 256 threads).
    // Prefetch each neighbor's 64-float feature row (global_prefetch_b8) so
    // the stage-3 gather hits cache.
    {
        int node = t >> 4;           // 0..15
        int k    = t & 15;           // 0..15
        int g = row0 + node;
        int j = nbr[(size_t)g * NKNN + k];
        Nb[node * NKNN + k] = j;
        __builtin_prefetch(&Xin[(size_t)j * 64], 0, 3);
        __builtin_prefetch(&Xin[(size_t)j * 64 + 32], 0, 3);
        float dx = means[(size_t)j * 2 + 0] - means[(size_t)g * 2 + 0];
        float dy = means[(size_t)j * 2 + 1] - means[(size_t)g * 2 + 1];
        float* e = &Emb[(node * NKNN + k) * EDIM];
        e[0] = 1.0f;
        float f = 1.0f;
#pragma unroll
        for (int q = 0; q < 5; ++q) {
            e[1 + 2 * q + 0] = __sinf(dx * f);
            e[1 + 2 * q + 1] = __sinf(dy * f);
            f *= 2.0f;
        }
    }

    // stage 1: H = tanh(X @ W1 + b1), one 16x16 column tile per wave (0..3)
    if (wave < 4) {
        int n0 = wave * 16;
        v8f c = {};
#pragma unroll
        for (int kb = 0; kb < 64; kb += 32) {
            v16h a = fragA_f32(Xin + (size_t)row0 * 64, 64, kb, lane);
            v16h b = fragB(Bt1, 64, n0, kb, lane);
            c = wmma_f16(a, b, c);
        }
        int n  = n0 + (lane & 15);
        int mb = (lane & 16) ? 8 : 0;
        float bias = b1[n];
#pragma unroll
        for (int r = 0; r < 8; ++r)
            Hh[(mb + r) * 64 + n] = (_Float16)fast_tanh(c[r] + bias);
    }
    __syncthreads();

    // stage 2: Tr = H @ W2 + b2, 44 column tiles across 8 waves.
    // A fragments are column-tile invariant: build once, stream B tiles.
    {
        v16h a0 = fragA_f16(Hh, 64, 0, lane);
        v16h a1 = fragA_f16(Hh, 64, 32, lane);
#pragma unroll
        for (int i = 0; i < 6; ++i) {
            int ct = wave + i * 8;
            if (ct >= 44) break;      // uniform per wave
            int n0 = ct * 16;
            v8f c = {};
            c = wmma_f16(a0, fragB(Bt2, 64, n0, 0, lane), c);
            c = wmma_f16(a1, fragB(Bt2, 64, n0, 32, lane), c);
            int n  = n0 + (lane & 15);
            int mb = (lane & 16) ? 8 : 0;
            float bias = b2[n];
#pragma unroll
            for (int r = 0; r < 8; ++r)
                Tr[(mb + r) * 704 + n] = c[r] + bias;
        }
    }
    __syncthreads();

    // stage 3: neighborhood aggregation (each thread: one node, 4 latents)
    {
        int node = t & 15;
        int l0   = t >> 4;            // 0..15
#pragma unroll
        for (int rr = 0; rr < 4; ++rr) {
            int l = l0 + rr * 16;
            const float* tr = &Tr[node * 704 + l * EDIM];
            float acc = 0.0f;
            for (int k = 0; k < NKNN; ++k) {
                const float* e = &Emb[(node * NKNN + k) * EDIM];
                float s = 0.0f;
#pragma unroll
                for (int q = 0; q < EDIM; ++q) s += tr[q] * e[q];
                acc += s * Xin[(size_t)Nb[node * NKNN + k] * 64 + l];
            }
            nfOut[(size_t)(row0 + node) * 64 + l] = acc * (1.0f / (float)NKNN);
        }
    }
}

// ---------------------------------------------------------------------------
// Kernel 6 (WMMA): head MLP on concat([feats, nf2]) (128->64->32->6),
// scatter 6 outputs into the 4 output sections.
// ---------------------------------------------------------------------------
__global__ __launch_bounds__(128) void deltas_kernel(
    const float* __restrict__ feats, const float* __restrict__ nf2,
    const _Float16* __restrict__ BtW1, const float* __restrict__ db1,
    const _Float16* __restrict__ BtW2, const float* __restrict__ db2,
    const float* __restrict__ W3, const float* __restrict__ db3,
    float* __restrict__ out, int N)
{
    __shared__ _Float16 H1[16 * 64];
    __shared__ float    H2[16 * 32];

    int row0 = blockIdx.x * 16;
    int t = threadIdx.x;
    int lane = t & 31, wave = t >> 5;

    // layer 1: 128 -> 64, 4 waves x 1 column tile; K blocks 0,32 come from
    // feats, blocks 64,96 from nf2 (the concat boundary is K-block aligned).
    {
        int n0 = wave * 16;
        const float* fbase = feats + (size_t)row0 * 64;
        const float* gbase = nf2   + (size_t)row0 * 64;
        v8f c = {};
        c = wmma_f16(fragA_f32(fbase, 64, 0,  lane), fragB(BtW1, 128, n0, 0,  lane), c);
        c = wmma_f16(fragA_f32(fbase, 64, 32, lane), fragB(BtW1, 128, n0, 32, lane), c);
        c = wmma_f16(fragA_f32(gbase, 64, 0,  lane), fragB(BtW1, 128, n0, 64, lane), c);
        c = wmma_f16(fragA_f32(gbase, 64, 32, lane), fragB(BtW1, 128, n0, 96, lane), c);
        int n  = n0 + (lane & 15);
        int mb = (lane & 16) ? 8 : 0;
        float bias = db1[n];
#pragma unroll
        for (int r = 0; r < 8; ++r)
            H1[(mb + r) * 64 + n] = (_Float16)fast_tanh(c[r] + bias);
    }
    __syncthreads();

    // layer 2: 64 -> 32, waves 0..1
    if (wave < 2) {
        int n0 = wave * 16;
        v8f c = {};
        c = wmma_f16(fragA_f16(H1, 64, 0,  lane), fragB(BtW2, 64, n0, 0,  lane), c);
        c = wmma_f16(fragA_f16(H1, 64, 32, lane), fragB(BtW2, 64, n0, 32, lane), c);
        int n  = n0 + (lane & 15);
        int mb = (lane & 16) ? 8 : 0;
        float bias = db2[n];
#pragma unroll
        for (int r = 0; r < 8; ++r)
            H2[(mb + r) * 32 + n] = fast_tanh(c[r] + bias);
    }
    __syncthreads();

    // layer 3: 32 -> 6, scalar scatter into output sections
    if (t < 96) {
        int node = t / 6, o = t % 6;
        float acc = db3[o];
#pragma unroll
        for (int j = 0; j < 32; ++j) acc += H2[node * 32 + j] * W3[j * 6 + o];
        int g = row0 + node;
        if (o < 2)       out[(size_t)g * 2 + o] = acc;                       // dmeans
        else if (o < 4)  out[(size_t)2 * N + (size_t)g * 2 + (o - 2)] = acc; // dscaling
        else if (o == 4) out[(size_t)4 * N + g] = acc;                       // dtransforms
        else             out[(size_t)5 * N + g] = acc;                       // du
    }
}

// ---------------------------------------------------------------------------
// Host-side orchestration.
// ---------------------------------------------------------------------------
extern "C" void kernel_launch(void* const* d_in, const int* in_sizes, int n_in,
                              void* d_out, int out_size, void* d_ws, size_t ws_size,
                              hipStream_t stream) {
    const int N = in_sizes[0] / 2;   // means is (N, 2)

    const float* means = (const float*)d_in[0];
    const float* cov   = (const float*)d_in[1];
    const float* u     = (const float*)d_in[2];
    const float* su    = (const float*)d_in[3];
    const float* sux   = (const float*)d_in[4];
    const float* suxx  = (const float*)d_in[5];
    const float* pde   = (const float*)d_in[6];
    const int*   nbr   = (const int*)d_in[7];

    // lt @ 8..13, tnets @ 14..43, ip @ 44..49, dtf @ 50..53, dn @ 54..59
    const float* ltW1 = (const float*)d_in[8];  const float* ltb1 = (const float*)d_in[9];
    const float* ltW2 = (const float*)d_in[10]; const float* ltb2 = (const float*)d_in[11];
    const float* ltW3 = (const float*)d_in[12]; const float* ltb3 = (const float*)d_in[13];

    TnetArgs ta;
    const int dds[5]   = {2, 1, 2, 2, 1};
    const int ooffs[5] = {0, 4, 5, 9, 13};
    for (int s = 0; s < 5; ++s) {
        int base = 14 + s * 6;
        ta.W1[s] = (const float*)d_in[base + 0]; ta.b1[s] = (const float*)d_in[base + 1];
        ta.W2[s] = (const float*)d_in[base + 2]; ta.b2[s] = (const float*)d_in[base + 3];
        ta.W3[s] = (const float*)d_in[base + 4]; ta.b3[s] = (const float*)d_in[base + 5];
        ta.dd[s] = dds[s]; ta.ooff[s] = ooffs[s];
    }

    const float* ipW1 = (const float*)d_in[44]; const float* ipb1 = (const float*)d_in[45];
    const float* ipW2 = (const float*)d_in[46]; const float* ipb2 = (const float*)d_in[47];
    const float* ipW3 = (const float*)d_in[48]; const float* ipb3 = (const float*)d_in[49];

    const float* dtfW1 = (const float*)d_in[50]; const float* dtfb1 = (const float*)d_in[51];
    const float* dtfW2 = (const float*)d_in[52]; const float* dtfb2 = (const float*)d_in[53];

    const float* dnW1 = (const float*)d_in[54]; const float* dnb1 = (const float*)d_in[55];
    const float* dnW2 = (const float*)d_in[56]; const float* dnb2 = (const float*)d_in[57];
    const float* dnW3 = (const float*)d_in[58]; const float* dnb3 = (const float*)d_in[59];

    // Workspace layout (float units)
    float* ws = (float*)d_ws;
    float* latent_sum = ws;                         // 64
    float* Tm         = ws + 64;                    // 16 (14 used)
    float* feats      = ws + 128;                   // N*64
    float* nf1        = feats + (size_t)N * 64;     // N*64
    float* nf2        = nf1   + (size_t)N * 64;     // N*64
    _Float16* hbase   = (_Float16*)(nf2 + (size_t)N * 64);
    _Float16* Bt_dtf1 = hbase;                      // 64*64
    _Float16* Bt_dtf2 = Bt_dtf1 + 64 * 64;          // 704*64
    _Float16* Bt_dn1  = Bt_dtf2 + 704 * 64;         // 64*128
    _Float16* Bt_dn2  = Bt_dn1  + 64 * 128;         // 32*64

    float* out = (float*)d_out;
    int tiles = N / 16;

    init_kernel<<<1, 64, 0, stream>>>(latent_sum);

    latent_kernel<<<512, 256, 0, stream>>>(means, cov, u, su, sux, suxx,
                                           ltW1, ltb1, ltW2, ltb2, ltW3, ltb3,
                                           latent_sum, N);

    tnet_kernel<<<1, 64, 0, stream>>>(latent_sum, 1.0f / (float)N, ta, Tm);

    feats_kernel<<<(N + 255) / 256, 256, 0, stream>>>(
        cov, u, su, sux, suxx, pde, Tm,
        ipW1, ipb1, ipW2, ipb2, ipW3, ipb3, feats, N);

    convert_weights_kernel<<<256, 256, 0, stream>>>(
        dtfW1, dtfW2, dnW1, dnW2, Bt_dtf1, Bt_dtf2, Bt_dn1, Bt_dn2);

    aggregate_kernel<<<tiles, 256, 0, stream>>>(
        feats, means, nbr, Bt_dtf1, dtfb1, Bt_dtf2, dtfb2, nf1);

    aggregate_kernel<<<tiles, 256, 0, stream>>>(
        nf1, means, nbr, Bt_dtf1, dtfb1, Bt_dtf2, dtfb2, nf2);

    deltas_kernel<<<tiles, 128, 0, stream>>>(
        feats, nf2, Bt_dn1, dnb1, Bt_dn2, dnb2, dnW3, dnb3, out, N);
}
